// HyperbolicStructureLearner_22479858827789
// MI455X (gfx1250) — matrix operand order repaired and checked by reading, not compiled, not run on hardware
//
#include <hip/hip_runtime.h>
#include <hip/hip_bf16.h>
#include <math.h>

#define N_NODES 50000
#define N_EDGES 800000
#define IN_DIM 128
#define HID_DIM 64
#define OUT_DIM 128
#define N_TILES (N_NODES / 16)   // 3125, exact
#define NEG_SLOPE 0.01f
#define EPS 1e-8f

typedef __attribute__((ext_vector_type(16))) _Float16 v16h;
typedef __attribute__((ext_vector_type(8)))  float    v8f;

static __device__ __forceinline__ v8f wmma_f32_f16(v16h a, v16h b, v8f c) {
  // (neg_a, A, neg_b, B, c_mod, C, reuse_a, reuse_b)
  return __builtin_amdgcn_wmma_f32_16x16x32_f16(false, a, false, b, (short)0, c, false, false);
}

// A fragment (16x32, M x K), 16-bit elements, contiguous-k layout (same as B, not ISA table order):
// lane = half*16 + m ; VGPR g : K = half*16 + 2g,(+1)
static __device__ __forceinline__ v16h load_afrag(const float* __restrict__ base,
                                                  int row, int ld, int k0, int half) {
  const float* p = base + (size_t)row * ld + k0;
  v16h f;
#pragma unroll
  for (int g = 0; g < 8; ++g) {
    int kk = half * 16 + 2 * g;
    float2 t = *(const float2*)(p + kk);
    f[2 * g]     = (_Float16)t.x;
    f[2 * g + 1] = (_Float16)t.y;
  }
  return f;
}

// B fragment (32x16, K x N) from row-major W[n][k]: lane = half*16 + n ;
// lanes 0-15 hold K=0..15, lanes 16-31 hold K=16..31 ; VGPR g : K = half*16 + 2g,(+1)
static __device__ __forceinline__ v16h load_bfrag(const float* __restrict__ base,
                                                  int nrow, int ld, int k0, int half) {
  const float* p = base + (size_t)nrow * ld + k0 + half * 16;
  v16h f;
#pragma unroll
  for (int g = 0; g < 8; ++g) {
    float2 t = *(const float2*)(p + 2 * g);
    f[2 * g]     = (_Float16)t.x;
    f[2 * g + 1] = (_Float16)t.y;
  }
  return f;
}

static __device__ __forceinline__ unsigned enc_f(float x) {
  unsigned u = __float_as_uint(x);
  return (u & 0x80000000u) ? ~u : (u | 0x80000000u);
}
static __device__ __forceinline__ float dec_f(unsigned k) {
  unsigned u = (k & 0x80000000u) ? (k & 0x7FFFFFFFu) : ~k;
  return __uint_as_float(u);
}

// ---------------- init: zero agg / smax_enc / ssum ----------------
__global__ void init_kernel(unsigned* __restrict__ p, int n) {
  int i = blockIdx.x * blockDim.x + threadIdx.x;
  if (i < n) p[i] = 0u;   // 0u == 0.0f == enc(-huge) lower bound
}

// ---------------- kernel 1: fused QKV GEMM + Lorentz norm + a/b scalars ----
__global__ void __launch_bounds__(128)
qkv_kernel(const float* __restrict__ xH, const float* __restrict__ xS,
           const float* __restrict__ Wq, const float* __restrict__ Wk,
           const float* __restrict__ Wv, const float* __restrict__ Ws,
           float* __restrict__ v_out, float* __restrict__ a_out,
           float* __restrict__ b_out) {
  __shared__ float lds[4][16 * 64];
  const int lane = threadIdx.x & 31;
  const int wave = threadIdx.x >> 5;
  const int tile = blockIdx.x * 4 + wave;
  if (tile >= N_TILES) return;
  const int row0 = tile * 16;
  const int r  = lane & 15;
  const int hh = lane >> 4;
  float* sm = lds[wave];

  for (int m = 0; m < 3; ++m) {
    const float* A = (m == 0) ? xS : xH;
    const float* W = (m == 0) ? Wq : ((m == 1) ? Wk : Wv);
    v8f acc0 = {}, acc1 = {}, acc2 = {}, acc3 = {};
#pragma unroll
    for (int kt = 0; kt < 4; ++kt) {
      v16h af = load_afrag(A, row0 + r, IN_DIM, kt * 32, hh);
      v16h b0 = load_bfrag(W,  0 + r, IN_DIM, kt * 32, hh);
      v16h b1 = load_bfrag(W, 16 + r, IN_DIM, kt * 32, hh);
      v16h b2 = load_bfrag(W, 32 + r, IN_DIM, kt * 32, hh);
      v16h b3 = load_bfrag(W, 48 + r, IN_DIM, kt * 32, hh);
      acc0 = wmma_f32_f16(af, b0, acc0);
      acc1 = wmma_f32_f16(af, b1, acc1);
      acc2 = wmma_f32_f16(af, b2, acc2);
      acc3 = wmma_f32_f16(af, b3, acc3);
    }
    // stage C tiles: VGPR g, lane -> (M = g + 8*hh, N = tile_n*16 + r)
#pragma unroll
    for (int g = 0; g < 8; ++g) {
      int Mr = g + 8 * hh;
      sm[Mr * 64 +      r] = acc0[g];
      sm[Mr * 64 + 16 + r] = acc1[g];
      sm[Mr * 64 + 32 + r] = acc2[g];
      sm[Mr * 64 + 48 + r] = acc3[g];
    }
    // Lorentz normalization: two lanes per row (hh halves of 32 cols)
    const float* hp = &sm[r * 64 + hh * 32];
    float ss = 0.f;
#pragma unroll
    for (int c = 0; c < 32; ++c) { float t = hp[c]; ss += t * t; }
    ss += __shfl_xor(ss, 16, 32);
    float h0 = sm[r * 64];
    float inner = ss - 2.f * h0 * h0;
    float scale = rsqrtf(fmaxf(fabsf(inner), EPS));

    if (m == 2) {
      float4* d4 = (float4*)&v_out[(size_t)(row0 + r) * HID_DIM + hh * 32];
#pragma unroll
      for (int c4 = 0; c4 < 8; ++c4) {
        float4 t = ((const float4*)hp)[c4];
        t.x *= scale; t.y *= scale; t.z *= scale; t.w *= scale;
        d4[c4] = t;
      }
    } else {
      const float* wseg = Ws + ((m == 0) ? 0 : HID_DIM) + hh * 32;
      float dot = 0.f;
#pragma unroll
      for (int c = 0; c < 32; ++c) dot += hp[c] * wseg[c];
      dot += __shfl_xor(dot, 16, 32);
      if (hh == 0) {
        float val = dot * scale;
        if (m == 0) a_out[row0 + r] = val;
        else        b_out[row0 + r] = val;
      }
    }
  }
}

// ---------------- kernel 2: per-edge score + segment max --------------------
__global__ void score_max_kernel(const int* __restrict__ src,
                                 const int* __restrict__ dst,
                                 const float* __restrict__ a,
                                 const float* __restrict__ b,
                                 unsigned* __restrict__ smax_enc) {
  int e = blockIdx.x * blockDim.x + threadIdx.x;
  if (e >= N_EDGES) return;
  int s = src[e], d = dst[e];
  float sc = a[s] + b[d];
  sc = (sc > 0.f) ? sc : NEG_SLOPE * sc;
  atomicMax(&smax_enc[s], enc_f(sc));
}

// ---------------- kernel 3: exp-weighted scatter (one wave per edge) --------
__global__ void __launch_bounds__(256)
scatter_kernel(const int* __restrict__ src, const int* __restrict__ dst,
               const float* __restrict__ a, const float* __restrict__ b,
               const unsigned* __restrict__ smax_enc,
               const float* __restrict__ v,
               float* __restrict__ ssum, float* __restrict__ agg) {
  int e    = (blockIdx.x * blockDim.x + threadIdx.x) >> 5;
  int lane = threadIdx.x & 31;
  if (e >= N_EDGES) return;
  int s = src[e], d = dst[e];
  float sc = a[s] + b[d];
  sc = (sc > 0.f) ? sc : NEG_SLOPE * sc;
  float w = __expf(sc - dec_f(smax_enc[s]));
  if (lane == 0) atomicAdd(&ssum[s], w);
  float v0 = v[(size_t)d * HID_DIM + lane];
  float v1 = v[(size_t)d * HID_DIM + lane + 32];
  atomicAdd(&agg[(size_t)s * HID_DIM + lane],      w * v0);
  atomicAdd(&agg[(size_t)s * HID_DIM + lane + 32], w * v1);
}

// ---------------- kernel 4: finalize (norm, proj GEMM, norm, +xH, norm) -----
__global__ void __launch_bounds__(128)
finalize_kernel(const float* __restrict__ xH, const float* __restrict__ Wproj,
                const float* __restrict__ agg, const float* __restrict__ ssum,
                float* __restrict__ z_out) {
  __shared__ float ldsA[4][16 * 64];
  __shared__ float ldsY[4][16 * 128];
  const int lane = threadIdx.x & 31;
  const int wave = threadIdx.x >> 5;
  const int tile = blockIdx.x * 4 + wave;
  if (tile >= N_TILES) return;
  const int row0 = tile * 16;
  const int r  = lane & 15;
  const int hh = lane >> 4;
  float* Abuf = ldsA[wave];
  float* Y    = ldsY[wave];

  // out = to_manifold(origin + agg/ssum) -> Abuf[16][64]
  {
    int row = row0 + r;
    float sden = ssum[row];
    float inv = (sden > 0.f) ? (1.f / sden) : 0.f;
    const float* ap = &agg[(size_t)row * HID_DIM + hh * 32];
    float vals[32];
    float ss = 0.f;
#pragma unroll
    for (int c = 0; c < 32; ++c) {
      float t = ap[c] * inv;
      if (hh == 0 && c == 0) t += 1.0f;   // Lorentz origin, 1/sqrt(K_H)
      vals[c] = t;
      ss += t * t;
    }
    ss += __shfl_xor(ss, 16, 32);
    float e0 = __shfl(vals[0], r, 32);    // lane r is (hh=0, row r)
    float inner = ss - 2.f * e0 * e0;
    float scale = rsqrtf(fmaxf(fabsf(inner), EPS));
#pragma unroll
    for (int c = 0; c < 32; ++c) Abuf[r * 64 + hh * 32 + c] = vals[c] * scale;
  }

  // y[16][128] = out[16][64] @ Wproj^T   (Wproj is [128][64] row-major)
  v8f acc[8];
#pragma unroll
  for (int j = 0; j < 8; ++j) acc[j] = (v8f){};
#pragma unroll
  for (int kt = 0; kt < 2; ++kt) {
    v16h af = load_afrag(Abuf, r, HID_DIM, kt * 32, hh);
#pragma unroll
    for (int j = 0; j < 8; ++j) {
      v16h bf = load_bfrag(Wproj, j * 16 + r, HID_DIM, kt * 32, hh);
      acc[j] = wmma_f32_f16(af, bf, acc[j]);
    }
  }
#pragma unroll
  for (int j = 0; j < 8; ++j)
#pragma unroll
    for (int g = 0; g < 8; ++g)
      Y[(g + 8 * hh) * 128 + j * 16 + r] = acc[j][g];

  // x = to_manifold(y) ; z = to_manifold(x + xH)
  {
    float* yp = &Y[r * 128 + hh * 64];
    float ss = 0.f;
#pragma unroll
    for (int c = 0; c < 64; ++c) { float t = yp[c]; ss += t * t; }
    ss += __shfl_xor(ss, 16, 32);
    float y0 = Y[r * 128];
    float scaleY = rsqrtf(fmaxf(fabsf(ss - 2.f * y0 * y0), EPS));

    const float* xhp = &xH[(size_t)(row0 + r) * OUT_DIM + hh * 64];
    float ss2 = 0.f;
#pragma unroll
    for (int c = 0; c < 64; ++c) {
      float t = yp[c] * scaleY + xhp[c];
      yp[c] = t;
      ss2 += t * t;
    }
    ss2 += __shfl_xor(ss2, 16, 32);
    float z0 = Y[r * 128];                 // same-wave DS ops are in-order
    float scaleZ = rsqrtf(fmaxf(fabsf(ss2 - 2.f * z0 * z0), EPS));

    float* zp = &z_out[(size_t)(row0 + r) * OUT_DIM + hh * 64];
#pragma unroll
    for (int c4 = 0; c4 < 16; ++c4) {
      float4 t = ((const float4*)yp)[c4];
      t.x *= scaleZ; t.y *= scaleZ; t.z *= scaleZ; t.w *= scaleZ;
      ((float4*)zp)[c4] = t;
    }
  }
}

extern "C" void kernel_launch(void* const* d_in, const int* in_sizes, int n_in,
                              void* d_out, int out_size, void* d_ws, size_t ws_size,
                              hipStream_t stream) {
  (void)in_sizes; (void)n_in; (void)out_size; (void)ws_size;
  const float* xH    = (const float*)d_in[0];
  const float* xS    = (const float*)d_in[1];
  const int*   ei    = (const int*)  d_in[2];
  const float* Wq    = (const float*)d_in[3];
  const float* Wk    = (const float*)d_in[4];
  const float* Wv    = (const float*)d_in[5];
  const float* Ws    = (const float*)d_in[6];
  const float* Wproj = (const float*)d_in[7];
  float* z_out = (float*)d_out;

  const int* src = ei;
  const int* dst = ei + N_EDGES;

  // workspace layout (floats): v[N*64] | a[N] | b[N] | agg[N*64] | smax[N] | ssum[N]
  float*    v_buf = (float*)d_ws;
  float*    a_buf = v_buf + (size_t)N_NODES * HID_DIM;
  float*    b_buf = a_buf + N_NODES;
  float*    agg   = b_buf + N_NODES;
  unsigned* smax  = (unsigned*)(agg + (size_t)N_NODES * HID_DIM);
  float*    ssum  = (float*)(smax + N_NODES);

  // zero agg + smax + ssum (contiguous region)
  {
    int n = N_NODES * HID_DIM + 2 * N_NODES;
    init_kernel<<<(n + 255) / 256, 256, 0, stream>>>((unsigned*)agg, n);
  }
  {
    int blocks = (N_TILES + 3) / 4;
    qkv_kernel<<<blocks, 128, 0, stream>>>(xH, xS, Wq, Wk, Wv, Ws,
                                           v_buf, a_buf, b_buf);
  }
  score_max_kernel<<<(N_EDGES + 255) / 256, 256, 0, stream>>>(src, dst, a_buf, b_buf, smax);
  scatter_kernel<<<(N_EDGES * 32 + 255) / 256, 256, 0, stream>>>(src, dst, a_buf, b_buf,
                                                                 smax, v_buf, ssum, agg);
  {
    int blocks = (N_TILES + 3) / 4;
    finalize_kernel<<<blocks, 128, 0, stream>>>(xH, Wproj, agg, ssum, z_out);
  }
}
